// _EGLayer_21964462751801
// MI455X (gfx1250) — matrix operand order repaired and checked by reading, not compiled, not run on hardware
//
#include <hip/hip_runtime.h>
#include <math.h>

#define FEAT 128
#define DFEAT 64
#define KIN 320          // 2*FEAT + DFEAT
#define R_CUTOFF 5.0f
#define ETILE 32         // edges per block (2 M-tiles of 16 per wave)
#define NTILE 32         // nodes per block

typedef __attribute__((ext_vector_type(16))) __bf16 v16bf;
typedef __attribute__((ext_vector_type(8)))  __bf16 v8bf;
typedef __attribute__((ext_vector_type(8)))  float  v8f;

__device__ __forceinline__ float silu_f(float v)    { return v * (1.0f / (1.0f + __expf(-v))); }
__device__ __forceinline__ float sigmoid_f(float v) { return 1.0f / (1.0f + __expf(-v)); }

__device__ __forceinline__ v16bf pack16(v8bf lo, v8bf hi) {
  v16bf r;
#pragma unroll
  for (int i = 0; i < 8; ++i) { r[i] = lo[i]; r[i + 8] = hi[i]; }
  return r;
}

// A-matrix fragment (16xK bf16 tile, row-major, row stride `stride` elems).
// ISA layout: lanes 0-15 -> M=lane, K in {0..7}u{16..23}; lanes 16-31 -> M=lane-16,
// K in {8..15}u{24..31}  (per k-step of 32).
__device__ __forceinline__ v16bf load_a_frag(const __bf16* tile, int stride, int k0, int lane) {
  int m  = lane & 15;
  int kh = (lane >> 4) << 3;
  const v8bf* p0 = (const v8bf*)(tile + m * stride + k0 + kh);
  const v8bf* p1 = (const v8bf*)(tile + m * stride + k0 + 16 + kh);
  return pack16(*p0, *p1);
}

// B-matrix fragment from bf16 weights, row-major KxN (N = ncols).
// ISA layout: lanes 0-15 -> N=lane, K in {0..7}u{16..23}; lanes 16-31 -> N=lane-16,
// K in {8..15}u{24..31}  (per k-step of 32).
__device__ __forceinline__ v16bf load_b_frag(const __bf16* W, int ncols, int k0, int n0, int lane) {
  int n  = n0 + (lane & 15);
  int kh = (lane >> 4) << 3;
  v16bf r;
#pragma unroll
  for (int i = 0; i < 8; ++i) {
    r[i]     = W[(size_t)(k0 + kh + i) * ncols + n];
    r[i + 8] = W[(size_t)(k0 + 16 + kh + i) * ncols + n];
  }
  return r;
}

#define WMMA_BF16(a, b, c) \
  __builtin_amdgcn_wmma_f32_16x16x32_bf16(false, (a), false, (b), (short)0, (c), false, false)

// ---------------------------------------------------------------- prep kernels
__global__ void cvt_f32_bf16_kernel(const float* __restrict__ s, __bf16* __restrict__ d, int n) {
  int i = blockIdx.x * 256 + threadIdx.x;
  if (i < n) d[i] = (__bf16)s[i];
}
__global__ void zero_f32_kernel(float* __restrict__ p, int n) {
  int i = blockIdx.x * 256 + threadIdx.x;
  if (i < n) p[i] = 0.0f;
}
__global__ void copy_f32_kernel(const float* __restrict__ s, float* __restrict__ d, int n) {
  int i = blockIdx.x * 256 + threadIdx.x;
  if (i < n) d[i] = s[i];
}

// ---------------------------------------------------------------- edge kernel
// One block = 32 edges, 8 waves; wave w computes output columns [16w, 16w+16)
// for both 16-edge M-tiles (B fragment reused across the pair of WMMAs).
__global__ __launch_bounds__(256)
void eg_edge_kernel(const float* __restrict__ h, const float* __restrict__ x,
                    const int* __restrict__ esrc, const int* __restrict__ edst,
                    const float* __restrict__ rbf_means, const float* __restrict__ rbf_stds,
                    const __bf16* __restrict__ W1, const float* __restrict__ b1,
                    const __bf16* __restrict__ W2, const float* __restrict__ b2,
                    const float* __restrict__ attw, const float* __restrict__ attb,
                    const __bf16* __restrict__ UXW1, const float* __restrict__ uxb1,
                    const float* __restrict__ uxw2,
                    float* __restrict__ node_msg, float* __restrict__ xout,
                    int E)
{
  __shared__ __align__(16) __bf16 sA[ETILE * KIN];     // edge feature tile (32 x 320)
  __shared__ __align__(16) __bf16 sAct[ETILE * FEAT];  // activations / gated messages
  __shared__ float sdir[ETILE][3];
  __shared__ float sdist[ETILE];
  __shared__ float smask[ETILE];
  __shared__ float sgate[ETILE];
  __shared__ float sred[ETILE];
  __shared__ int   ssrc[ETILE];
  __shared__ int   sdst[ETILE];

  const int tid = threadIdx.x;
  const int e0  = blockIdx.x * ETILE;

  // ---- per-edge geometry
  if (tid < ETILE) {
    int e = e0 + tid;
    bool valid = e < E;
    int s = valid ? esrc[e] : 0;
    int d = valid ? edst[e] : 0;
    ssrc[tid] = s; sdst[tid] = d;
    float dx = x[d * 3 + 0] - x[s * 3 + 0];
    float dy = x[d * 3 + 1] - x[s * 3 + 1];
    float dz = x[d * 3 + 2] - x[s * 3 + 2];
    float dist = sqrtf(fmaxf(dx * dx + dy * dy + dz * dz, 1e-24f));
    float inv = 1.0f / dist;
    sdir[tid][0] = dx * inv; sdir[tid][1] = dy * inv; sdir[tid][2] = dz * inv;
    sdist[tid] = dist;
    smask[tid] = (valid && dist < R_CUTOFF) ? 1.0f : 0.0f;
  }
  __syncthreads();

  // ---- stage edge input rows: [h_src | h_dst | rbf] as bf16
#pragma unroll
  for (int half = 0; half < 2; ++half) {
    int er = half * 16 + (tid >> 4);   // edge row 0..31
    int c  = (tid & 15) * 8;           // feature chunk
    int s = ssrc[er], d = sdst[er];
    const float* hs = h + (size_t)s * FEAT + c;
    const float* hd = h + (size_t)d * FEAT + c;
#pragma unroll
    for (int i = 0; i < 8; ++i) sA[er * KIN + c + i] = (__bf16)hs[i];
#pragma unroll
    for (int i = 0; i < 8; ++i) sA[er * KIN + FEAT + c + i] = (__bf16)hd[i];
    if ((tid & 15) < 8) {
      int j0 = (tid & 7) * 8;
      float dist = sdist[er];
#pragma unroll
      for (int i = 0; i < 8; ++i) {
        float t = (dist - rbf_means[j0 + i]) / rbf_stds[j0 + i];
        sA[er * KIN + 2 * FEAT + j0 + i] = (__bf16)__expf(-0.5f * t * t);
      }
    }
  }
  __syncthreads();

  const int wid  = tid >> 5;
  const int lane = tid & 31;
  const int n0   = wid * 16;
  const int ncol = n0 + (lane & 15);
  const int moff = (lane >> 4) * 8;   // C/D row offset: VGPR r holds M = r + moff

  // ---- GEMM1: (32x320) x (320x128); B fragment shared by both M-tiles
  v8f acc1a = {}, acc1b = {};
#pragma unroll
  for (int kt = 0; kt < KIN / 32; ++kt) {
    v16bf b  = load_b_frag(W1, FEAT, kt * 32, n0, lane);
    v16bf a0 = load_a_frag(sA,             KIN, kt * 32, lane);
    v16bf a1 = load_a_frag(sA + 16 * KIN,  KIN, kt * 32, lane);
    acc1a = WMMA_BF16(a0, b, acc1a);
    acc1b = WMMA_BF16(a1, b, acc1b);
  }
  {
    float bias = b1[ncol];
#pragma unroll
    for (int r = 0; r < 8; ++r) {
      sAct[(r + moff) * FEAT + ncol]        = (__bf16)silu_f(acc1a[r] + bias);
      sAct[(16 + r + moff) * FEAT + ncol]   = (__bf16)silu_f(acc1b[r] + bias);
    }
  }
  __syncthreads();

  // ---- GEMM2: (32x128) x (128x128)
  v8f acc2a = {}, acc2b = {};
#pragma unroll
  for (int kt = 0; kt < FEAT / 32; ++kt) {
    v16bf b  = load_b_frag(W2, FEAT, kt * 32, n0, lane);
    v16bf a0 = load_a_frag(sAct,             FEAT, kt * 32, lane);
    v16bf a1 = load_a_frag(sAct + 16 * FEAT, FEAT, kt * 32, lane);
    acc2a = WMMA_BF16(a0, b, acc2a);
    acc2b = WMMA_BF16(a1, b, acc2b);
  }
  float mreg[2][8];
  {
    float bias = b2[ncol];
#pragma unroll
    for (int r = 0; r < 8; ++r) {
      mreg[0][r] = silu_f(acc2a[r] + bias);
      mreg[1][r] = silu_f(acc2b[r] + bias);
    }
  }

  // ---- attention gate: att = sigmoid(m . attw + attb), reduce columns via LDS
  if (tid < ETILE) sred[tid] = 0.0f;
  __syncthreads();
  {
    float aw = attw[ncol];
#pragma unroll
    for (int t = 0; t < 2; ++t)
#pragma unroll
      for (int r = 0; r < 8; ++r)
        atomicAdd(&sred[t * 16 + r + moff], mreg[t][r] * aw);
  }
  __syncthreads();
  if (tid < ETILE) sgate[tid] = smask[tid] * sigmoid_f(sred[tid] + attb[0]);
  __syncthreads();

  // ---- gate, scatter to node_msg, stash gated m (bf16) for ux path
#pragma unroll
  for (int t = 0; t < 2; ++t) {
#pragma unroll
    for (int r = 0; r < 8; ++r) {
      int M = t * 16 + r + moff;
      float g = mreg[t][r] * sgate[M];
      atomicAdd(&node_msg[(size_t)sdst[M] * FEAT + ncol], g);
      sAct[M * FEAT + ncol] = (__bf16)g;
    }
  }
  __syncthreads();

  // ---- GEMM3 (ux path): (32x128) x (128x128) -> silu -> dot with uxw2
  v8f acc3a = {}, acc3b = {};
#pragma unroll
  for (int kt = 0; kt < FEAT / 32; ++kt) {
    v16bf b  = load_b_frag(UXW1, FEAT, kt * 32, n0, lane);
    v16bf a0 = load_a_frag(sAct,             FEAT, kt * 32, lane);
    v16bf a1 = load_a_frag(sAct + 16 * FEAT, FEAT, kt * 32, lane);
    acc3a = WMMA_BF16(a0, b, acc3a);
    acc3b = WMMA_BF16(a1, b, acc3b);
  }
  if (tid < ETILE) sred[tid] = 0.0f;
  __syncthreads();
  {
    float bias = uxb1[ncol];
    float w2   = uxw2[ncol];
#pragma unroll
    for (int r = 0; r < 8; ++r) {
      atomicAdd(&sred[r + moff],      silu_f(acc3a[r] + bias) * w2);
      atomicAdd(&sred[16 + r + moff], silu_f(acc3b[r] + bias) * w2);
    }
  }
  __syncthreads();

  // ---- displacement scatter
  if (tid < ETILE) {
    int e = e0 + tid;
    if (e < E) {
      float mag = tanhf(sred[tid]) * smask[tid];   // SPEED_FACTOR == 1
      int d = sdst[tid];
      atomicAdd(&xout[(size_t)d * 3 + 0], mag * sdir[tid][0]);
      atomicAdd(&xout[(size_t)d * 3 + 1], mag * sdir[tid][1]);
      atomicAdd(&xout[(size_t)d * 3 + 2], mag * sdir[tid][2]);
    }
  }
}

// ---------------------------------------------------------------- node kernel
// h_out = h + (silu([h | node_msg] @ uh_w1 + b) @ uh_w2 + b)
__global__ __launch_bounds__(256)
void eg_node_kernel(const float* __restrict__ h, const float* __restrict__ node_msg,
                    const __bf16* __restrict__ UHW1, const float* __restrict__ uhb1,
                    const __bf16* __restrict__ UHW2, const float* __restrict__ uhb2,
                    float* __restrict__ hout, int N)
{
  __shared__ __align__(16) __bf16 sA[NTILE * 256];
  __shared__ __align__(16) __bf16 sAct[NTILE * FEAT];
  const int tid = threadIdx.x;
  const int nb  = blockIdx.x * NTILE;

#pragma unroll
  for (int half = 0; half < 2; ++half) {
    int row = half * 16 + (tid >> 4);
    int c   = (tid & 15) * 8;
    int node = nb + row; if (node >= N) node = N - 1;
    const float* hp = h        + (size_t)node * FEAT + c;
    const float* mp = node_msg + (size_t)node * FEAT + c;
#pragma unroll
    for (int i = 0; i < 8; ++i) sA[row * 256 + c + i]        = (__bf16)hp[i];
#pragma unroll
    for (int i = 0; i < 8; ++i) sA[row * 256 + FEAT + c + i] = (__bf16)mp[i];
  }
  __syncthreads();

  const int wid  = tid >> 5;
  const int lane = tid & 31;
  const int n0   = wid * 16;
  const int ncol = n0 + (lane & 15);
  const int moff = (lane >> 4) * 8;

  v8f acc1a = {}, acc1b = {};
#pragma unroll
  for (int kt = 0; kt < 256 / 32; ++kt) {
    v16bf b  = load_b_frag(UHW1, FEAT, kt * 32, n0, lane);
    v16bf a0 = load_a_frag(sA,            256, kt * 32, lane);
    v16bf a1 = load_a_frag(sA + 16 * 256, 256, kt * 32, lane);
    acc1a = WMMA_BF16(a0, b, acc1a);
    acc1b = WMMA_BF16(a1, b, acc1b);
  }
  {
    float bias = uhb1[ncol];
#pragma unroll
    for (int r = 0; r < 8; ++r) {
      sAct[(r + moff) * FEAT + ncol]      = (__bf16)silu_f(acc1a[r] + bias);
      sAct[(16 + r + moff) * FEAT + ncol] = (__bf16)silu_f(acc1b[r] + bias);
    }
  }
  __syncthreads();

  v8f acc2a = {}, acc2b = {};
#pragma unroll
  for (int kt = 0; kt < FEAT / 32; ++kt) {
    v16bf b  = load_b_frag(UHW2, FEAT, kt * 32, n0, lane);
    v16bf a0 = load_a_frag(sAct,             FEAT, kt * 32, lane);
    v16bf a1 = load_a_frag(sAct + 16 * FEAT, FEAT, kt * 32, lane);
    acc2a = WMMA_BF16(a0, b, acc2a);
    acc2b = WMMA_BF16(a1, b, acc2b);
  }
  {
    float bias = uhb2[ncol];
#pragma unroll
    for (int r = 0; r < 8; ++r) {
      int M0 = nb + r + moff;
      int M1 = nb + 16 + r + moff;
      if (M0 < N)
        hout[(size_t)M0 * FEAT + ncol] = h[(size_t)M0 * FEAT + ncol] + acc2a[r] + bias;
      if (M1 < N)
        hout[(size_t)M1 * FEAT + ncol] = h[(size_t)M1 * FEAT + ncol] + acc2b[r] + bias;
    }
  }
}

// ---------------------------------------------------------------- launch
extern "C" void kernel_launch(void* const* d_in, const int* in_sizes, int n_in,
                              void* d_out, int out_size, void* d_ws, size_t ws_size,
                              hipStream_t stream) {
  const float* h    = (const float*)d_in[0];
  const float* x    = (const float*)d_in[1];
  const int*   esrc = (const int*)d_in[2];
  const int*   edst = (const int*)d_in[3];
  const float* rbfm = (const float*)d_in[4];
  const float* rbfs = (const float*)d_in[5];
  const float* w1   = (const float*)d_in[6];
  const float* b1   = (const float*)d_in[7];
  const float* w2   = (const float*)d_in[8];
  const float* b2   = (const float*)d_in[9];
  const float* attw = (const float*)d_in[10];
  const float* attb = (const float*)d_in[11];
  const float* uxw1 = (const float*)d_in[12];
  const float* uxb1 = (const float*)d_in[13];
  const float* uxw2 = (const float*)d_in[14];
  const float* uhw1 = (const float*)d_in[15];
  const float* uhb1 = (const float*)d_in[16];
  const float* uhw2 = (const float*)d_in[17];
  const float* uhb2 = (const float*)d_in[18];

  const int E = in_sizes[2];
  const int N = in_sizes[0] / FEAT;

  // workspace layout (16B-aligned regions)
  char* ws = (char*)d_ws;
  __bf16* W1bf   = (__bf16*)(ws + 0);        // 320*128 bf16  = 81920 B
  __bf16* W2bf   = (__bf16*)(ws + 81920);    // 128*128 bf16  = 32768 B
  __bf16* UXW1bf = (__bf16*)(ws + 114688);   // 128*128 bf16
  __bf16* UHW1bf = (__bf16*)(ws + 147456);   // 256*128 bf16  = 65536 B
  __bf16* UHW2bf = (__bf16*)(ws + 212992);   // 128*128 bf16
  float*  node_msg = (float*)(ws + 245760);  // N*128 f32 accumulator

  float* hout = (float*)d_out;
  float* xout = hout + (size_t)N * FEAT;

  // weight conversion f32 -> bf16
  cvt_f32_bf16_kernel<<<(KIN * FEAT + 255) / 256, 256, 0, stream>>>(w1, W1bf, KIN * FEAT);
  cvt_f32_bf16_kernel<<<(FEAT * FEAT + 255) / 256, 256, 0, stream>>>(w2, W2bf, FEAT * FEAT);
  cvt_f32_bf16_kernel<<<(FEAT * FEAT + 255) / 256, 256, 0, stream>>>(uxw1, UXW1bf, FEAT * FEAT);
  cvt_f32_bf16_kernel<<<(2 * FEAT * FEAT + 255) / 256, 256, 0, stream>>>(uhw1, UHW1bf, 2 * FEAT * FEAT);
  cvt_f32_bf16_kernel<<<(FEAT * FEAT + 255) / 256, 256, 0, stream>>>(uhw2, UHW2bf, FEAT * FEAT);

  // init accumulators / outputs
  zero_f32_kernel<<<(N * FEAT + 255) / 256, 256, 0, stream>>>(node_msg, N * FEAT);
  copy_f32_kernel<<<(N * 3 + 255) / 256, 256, 0, stream>>>(x, xout, N * 3);

  // fused edge pipeline
  eg_edge_kernel<<<(E + ETILE - 1) / ETILE, 256, 0, stream>>>(
      h, x, esrc, edst, rbfm, rbfs,
      W1bf, b1, W2bf, b2, attw, attb,
      UXW1bf, uxb1, uxw2, node_msg, xout, E);

  // node update
  eg_node_kernel<<<(N + NTILE - 1) / NTILE, 256, 0, stream>>>(
      h, node_msg, UHW1bf, uhb1, UHW2bf, uhb2, hout, N);
}
